// PointTransformerLayer_87187836109558
// MI455X (gfx1250) — matrix operand order repaired and checked by reading, not compiled, not run on hardware
//
#include <hip/hip_runtime.h>
#include <hip/hip_bf16.h>

#define BB   2
#define NPTS 8192
#define DD   512
#define KNN  16
#define MM   (BB * NPTS)      // 16384
#define CHUNK 2048            // knn LDS tile (24 KB)

typedef __attribute__((ext_vector_type(16))) __bf16 v16bf;
typedef __attribute__((ext_vector_type(8)))  __bf16 v8bf;
typedef __attribute__((ext_vector_type(4)))  __bf16 v4bf;
typedef __attribute__((ext_vector_type(8)))  float  v8f;
typedef __attribute__((ext_vector_type(4)))  float  v4f;
typedef __attribute__((ext_vector_type(4)))  int    v4i;

#define AS1 __attribute__((address_space(1)))
#define AS3 __attribute__((address_space(3)))

#define WMMA_BF16(a, b, c) \
    __builtin_amdgcn_wmma_f32_16x16x32_bf16(false, (a), false, (b), (short)0, (c), false, false)

// ---------------------------------------------------------------------------
// f32 -> bf16 conversion of x (4 elements / thread, vectorized)
// ---------------------------------------------------------------------------
__global__ void cvt_x_kernel(const float* __restrict__ x, __bf16* __restrict__ xb) {
    size_t i = (size_t)blockIdx.x * blockDim.x + threadIdx.x;   // 0 .. MM*DD/4
    v4f v = ((const v4f*)x)[i];
    v4bf o;
    o[0] = (__bf16)v[0]; o[1] = (__bf16)v[1];
    o[2] = (__bf16)v[2]; o[3] = (__bf16)v[3];
    ((v4bf*)xb)[i] = o;
}

// convert W_v (rows 1024..1535 of W_qkv) and W_fc to bf16
__global__ void cvt_w_kernel(const float* __restrict__ Wqkv, const float* __restrict__ Wfc,
                             __bf16* __restrict__ wvB, __bf16* __restrict__ wfB) {
    size_t i = (size_t)blockIdx.x * blockDim.x + threadIdx.x;   // 0 .. 512*512
    wvB[i] = (__bf16)Wqkv[(size_t)2 * DD * DD + i];
    wfB[i] = (__bf16)Wfc[i];
}

// ---------------------------------------------------------------------------
// rowsum(W_q)[d], rowsum(W_k)[d] and bias sums (q.sum / k.sum collapse)
// ---------------------------------------------------------------------------
__global__ void rowsum_kernel(const float* __restrict__ Wqkv, const float* __restrict__ bqkv,
                              float* __restrict__ wq, float* __restrict__ wk,
                              float* __restrict__ bsum) {
    int d = blockIdx.x * blockDim.x + threadIdx.x;   // 0..511
    float sq = 0.f, sk = 0.f;
    for (int e = 0; e < DD; ++e) {
        sq += Wqkv[(size_t)e * DD + d];
        sk += Wqkv[(size_t)(DD + e) * DD + d];
    }
    wq[d] = sq;
    wk[d] = sk;
    if (d == 0) {
        float b0 = 0.f, b1 = 0.f;
        for (int e = 0; e < DD; ++e) { b0 += bqkv[e]; b1 += bqkv[DD + e]; }
        bsum[0] = b0; bsum[1] = b1;
    }
}

// ---------------------------------------------------------------------------
// qs[p] = x[p]·wq + bsum_q ; ks[p] = x[p]·wk + bsum_k   (one wave32 / point)
// ---------------------------------------------------------------------------
__global__ void qsks_kernel(const float* __restrict__ x,
                            const float* __restrict__ wq, const float* __restrict__ wk,
                            const float* __restrict__ bsum,
                            float* __restrict__ qs, float* __restrict__ ks) {
    const int lane = threadIdx.x & 31;
    const int wid  = threadIdx.x >> 5;
    const int p    = blockIdx.x * 8 + wid;            // 0..MM-1
    const float* xr = x + (size_t)p * DD;
    float sq = 0.f, sk = 0.f;
    for (int d = lane; d < DD; d += 32) {
        float xv = xr[d];
        sq += xv * wq[d];
        sk += xv * wk[d];
    }
    #pragma unroll
    for (int off = 16; off; off >>= 1) {
        sq += __shfl_xor(sq, off, 32);
        sk += __shfl_xor(sk, off, 32);
    }
    if (lane == 0) { qs[p] = sq + bsum[0]; ks[p] = sk + bsum[1]; }
}

// ---------------------------------------------------------------------------
// KNN: one thread per query point, xyz streamed through LDS in 24KB chunks,
// register-resident sorted top-16 (order irrelevant downstream).
// ---------------------------------------------------------------------------
__global__ void knn_kernel(const float* __restrict__ xyz, int* __restrict__ knn) {
    __shared__ float sx[CHUNK * 3];
    const int b  = blockIdx.x / (NPTS / 256);
    const int qn = (blockIdx.x % (NPTS / 256)) * 256 + threadIdx.x;
    const float* base = xyz + (size_t)b * NPTS * 3;
    const float qx = base[qn * 3 + 0];
    const float qy = base[qn * 3 + 1];
    const float qz = base[qn * 3 + 2];

    float bd[KNN]; int bi[KNN];
    #pragma unroll
    for (int j = 0; j < KNN; ++j) { bd[j] = 3.4e38f; bi[j] = 0; }

    for (int c = 0; c < NPTS; c += CHUNK) {
        __syncthreads();
        for (int i = threadIdx.x; i < CHUNK * 3; i += 256)
            sx[i] = base[(size_t)c * 3 + i];
        __syncthreads();
        for (int m = 0; m < CHUNK; ++m) {
            float dx = sx[m * 3 + 0] - qx;
            float dy = sx[m * 3 + 1] - qy;
            float dz = sx[m * 3 + 2] - qz;
            float d2 = dx * dx + dy * dy + dz * dz;
            if (d2 < bd[KNN - 1]) {
                float d = d2; int id = c + m;
                #pragma unroll
                for (int j = 0; j < KNN; ++j) {
                    bool sw = d < bd[j];
                    float td = sw ? bd[j] : d;
                    int   ti = sw ? bi[j] : id;
                    bd[j] = sw ? d  : bd[j];
                    bi[j] = sw ? id : bi[j];
                    d = td; id = ti;
                }
            }
        }
    }
    int* o = knn + ((size_t)b * NPTS + qn) * KNN;
    #pragma unroll
    for (int j = 0; j < KNN; ++j) o[j] = bi[j];
}

// ---------------------------------------------------------------------------
// bf16 WMMA GEMM: C[M x 512] = A[M x 512] * B[512 x 512]^T (B stored N-major).
// Block = 8 waves, block tile 256(M) x 32(N); wave tile 32x32 (4 accumulators,
// 2 A-frags x 2 B-frags per k-step). The 32-col B tile over full K is one
// contiguous 32KB blob -> staged once per block into LDS via the gfx1250
// async global->LDS path (ASYNCcnt), then read as ds_load_b128 fragments.
// FC=false: out_bf = C + bias (bf16 V matrix)
// FC=true : out_f  = C + bias + residual (final f32 output)
// ---------------------------------------------------------------------------
template <bool FC>
__global__ void gemm_wmma_kernel(const __bf16* __restrict__ A, const __bf16* __restrict__ Bw,
                                 const float* __restrict__ bias, const float* __restrict__ resid,
                                 float* __restrict__ outF, __bf16* __restrict__ outBf) {
    __shared__ __align__(16) __bf16 sB[32 * DD];   // 32 KB
    const int lane = threadIdx.x & 31;
    const int wid  = threadIdx.x >> 5;
    const int mb   = blockIdx.x >> 4;              // 64 m-blocks of 256 rows
    const int nb   = blockIdx.x & 15;              // 16 n-blocks of 32 cols
    const int n0   = nb << 5;
    const int m0   = (mb << 8) + (wid << 5);       // 32 rows per wave

    // ---- stage B tile (rows n0..n0+31, all K: contiguous 32KB) into LDS ----
    {
        const __bf16* gsrc = Bw + (size_t)n0 * DD;
#if __has_builtin(__builtin_amdgcn_global_load_async_to_lds_b128)
        #pragma unroll
        for (int i = 0; i < 8; ++i) {
            const int c = i * 256 + (int)threadIdx.x;   // 16-byte chunk id
            __builtin_amdgcn_global_load_async_to_lds_b128(
                (AS1 v4i*)(uintptr_t)(gsrc + c * 8),
                (AS3 v4i*)(uintptr_t)(sB + c * 8),
                0, 0);
        }
#if __has_builtin(__builtin_amdgcn_s_wait_asynccnt)
        __builtin_amdgcn_s_wait_asynccnt(0);
#else
        asm volatile("s_wait_asynccnt 0" ::: "memory");
#endif
#else
        #pragma unroll
        for (int i = 0; i < 8; ++i) {
            const int c = i * 256 + (int)threadIdx.x;
            ((v8bf*)sB)[c] = ((const v8bf*)gsrc)[c];
        }
#endif
        __syncthreads();
    }

    const int r  = lane & 15;
    const int hi = lane >> 4;

    // ISA bf16 A layout: lane-half hi=0 holds K {0..7,16..23}, hi=1 holds {8..15,24..31}
    const __bf16* ap0 = A + (size_t)(m0 + r) * DD + hi * 8;
    const __bf16* ap1 = A + (size_t)(m0 + 16 + r) * DD + hi * 8;
    // ISA bf16 B layout: lane = output column, lane-half hi=0 holds K {0..7,16..23}, hi=1 holds {8..15,24..31} (LDS)
    const __bf16* lb0 = sB + (size_t)r * DD + hi * 8;
    const __bf16* lb1 = sB + (size_t)(16 + r) * DD + hi * 8;

    v8f c00 = {}, c01 = {}, c10 = {}, c11 = {};
    for (int k0 = 0; k0 < DD; k0 += 32) {
        __builtin_prefetch(ap0 + k0 + 128, 0, 3);
        v16bf a0, a1, b0, b1;
        ((v8bf*)&a0)[0] = *(const v8bf*)(ap0 + k0);
        ((v8bf*)&a0)[1] = *(const v8bf*)(ap0 + k0 + 16);
        ((v8bf*)&a1)[0] = *(const v8bf*)(ap1 + k0);
        ((v8bf*)&a1)[1] = *(const v8bf*)(ap1 + k0 + 16);
        ((v8bf*)&b0)[0] = *(const v8bf*)(lb0 + k0);
        ((v8bf*)&b0)[1] = *(const v8bf*)(lb0 + k0 + 16);
        ((v8bf*)&b1)[0] = *(const v8bf*)(lb1 + k0);
        ((v8bf*)&b1)[1] = *(const v8bf*)(lb1 + k0 + 16);
        c00 = WMMA_BF16(a0, b0, c00);
        c01 = WMMA_BF16(a0, b1, c01);
        c10 = WMMA_BF16(a1, b0, c10);
        c11 = WMMA_BF16(a1, b1, c11);
    }

    // C/D layout: lane gives N=lane%16, M-half = lane/16; VGPR rr gives M offset
    #pragma unroll
    for (int rr = 0; rr < 8; ++rr) {
        const int rowA = m0 + hi * 8 + rr;
        const int rowB = rowA + 16;
        const int col0 = n0 + r;
        const int col1 = n0 + 16 + r;
        if (FC) {
            outF[(size_t)rowA * DD + col0] = c00[rr] + bias[col0] + resid[(size_t)rowA * DD + col0];
            outF[(size_t)rowA * DD + col1] = c01[rr] + bias[col1] + resid[(size_t)rowA * DD + col1];
            outF[(size_t)rowB * DD + col0] = c10[rr] + bias[col0] + resid[(size_t)rowB * DD + col0];
            outF[(size_t)rowB * DD + col1] = c11[rr] + bias[col1] + resid[(size_t)rowB * DD + col1];
        } else {
            outBf[(size_t)rowA * DD + col0] = (__bf16)(c00[rr] + bias[col0]);
            outBf[(size_t)rowA * DD + col1] = (__bf16)(c01[rr] + bias[col1]);
            outBf[(size_t)rowB * DD + col0] = (__bf16)(c10[rr] + bias[col0]);
            outBf[(size_t)rowB * DD + col1] = (__bf16)(c11[rr] + bias[col1]);
        }
    }
}

// ---------------------------------------------------------------------------
// Scalar attention: softmax over 16 neighbor logits, weighted gather of V.
// One wave32 per point; each lane owns 16 contiguous channels.
// ---------------------------------------------------------------------------
__global__ void attn_kernel(const int* __restrict__ knn,
                            const float* __restrict__ qs, const float* __restrict__ ks,
                            const __bf16* __restrict__ vbf, __bf16* __restrict__ xout) {
    const int lane = threadIdx.x & 31;
    const int wid  = threadIdx.x >> 5;
    const int p    = blockIdx.x * 8 + wid;        // 0..MM-1
    const int b    = p / NPTS;
    const int* idx = knn + (size_t)p * KNN;
    const float qv = qs[p];
    const float scale = 0.044194173824159216f;    // 1/sqrt(512)

    int   id[KNN];
    float l[KNN];
    #pragma unroll
    for (int k = 0; k < KNN; ++k) {
        id[k] = b * NPTS + idx[k];
        l[k]  = qv * ks[id[k]] * scale;
    }
    float mx = l[0];
    #pragma unroll
    for (int k = 1; k < KNN; ++k) mx = fmaxf(mx, l[k]);
    float s = 0.f;
    #pragma unroll
    for (int k = 0; k < KNN; ++k) { l[k] = __expf(l[k] - mx); s += l[k]; }
    const float inv = 1.0f / s;

    float acc[16];
    #pragma unroll
    for (int j = 0; j < 16; ++j) acc[j] = 0.f;
    const int doff = lane * 16;
    for (int k = 0; k < KNN; ++k) {
        const float w = l[k] * inv;
        const v8bf* vp = (const v8bf*)(vbf + (size_t)id[k] * DD + doff);
        v8bf v0 = vp[0], v1 = vp[1];
        #pragma unroll
        for (int j = 0; j < 8; ++j) {
            acc[j]     += w * (float)v0[j];
            acc[8 + j] += w * (float)v1[j];
        }
    }
    v8bf o0, o1;
    #pragma unroll
    for (int j = 0; j < 8; ++j) { o0[j] = (__bf16)acc[j]; o1[j] = (__bf16)acc[8 + j]; }
    v8bf* op = (v8bf*)(xout + (size_t)p * DD + doff);
    op[0] = o0; op[1] = o1;
}

// ---------------------------------------------------------------------------
extern "C" void kernel_launch(void* const* d_in, const int* in_sizes, int n_in,
                              void* d_out, int out_size, void* d_ws, size_t ws_size,
                              hipStream_t stream) {
    const float* x    = (const float*)d_in[0];
    const float* xyz  = (const float*)d_in[1];
    const float* Wqkv = (const float*)d_in[2];
    const float* bqkv = (const float*)d_in[3];
    const float* Wfc  = (const float*)d_in[4];
    const float* bfc  = (const float*)d_in[5];
    float* out = (float*)d_out;

    char* ws = (char*)d_ws;
    __bf16* xb   = (__bf16*)(ws);                                  // 16 MB
    __bf16* wvB  = (__bf16*)(ws + (size_t)16777216);               // 512 KB
    __bf16* wfB  = (__bf16*)(ws + (size_t)16777216 + 524288);      // 512 KB
    float*  wq   = (float*) (ws + (size_t)17825792);               // 2 KB
    float*  wk   = wq + DD;
    float*  bsum = wk + DD;                                        // 2 floats
    float*  qs   = (float*) (ws + (size_t)17825792 + 8192);        // 64 KB
    float*  ks   = qs + MM;                                        // 64 KB
    int*    knn  = (int*)   (ws + (size_t)17825792 + 8192 + 131072);   // 1 MB
    __bf16* vb   = (__bf16*)(ws + (size_t)17825792 + 8192 + 131072 + 1048576); // 16 MB
    __bf16* xo   = vb + (size_t)MM * DD;                           // 16 MB

    cvt_x_kernel  <<<(MM * DD / 4) / 256, 256, 0, stream>>>(x, xb);
    cvt_w_kernel  <<<(DD * DD) / 256, 256, 0, stream>>>(Wqkv, Wfc, wvB, wfB);
    rowsum_kernel <<<DD / 256, 256, 0, stream>>>(Wqkv, bqkv, wq, wk, bsum);
    qsks_kernel   <<<MM / 8, 256, 0, stream>>>(x, wq, wk, bsum, qs, ks);
    knn_kernel    <<<BB * (NPTS / 256), 256, 0, stream>>>(xyz, knn);

    const int gemm_blocks = (MM / 256) * (DD / 32);                // 64 * 16 = 1024
    gemm_wmma_kernel<false><<<gemm_blocks, 256, 0, stream>>>(xb, wvB, bqkv + 2 * DD, nullptr, nullptr, vb);
    attn_kernel   <<<MM / 8, 256, 0, stream>>>(knn, qs, ks, vb, xo);
    gemm_wmma_kernel<true> <<<gemm_blocks, 256, 0, stream>>>(xo, wfB, bfc, x, out, nullptr);
}
